// Net_60533269070095
// MI455X (gfx1250) — matrix operand order repaired and checked by reading, not compiled, not run on hardware
//
#include <hip/hip_runtime.h>

// ---------------- problem constants ----------------
#define KTAPS 5
#define NB    100
#define NBPAD 112        // 7 M-tiles of 16 for fc1
#define N0    78400      // 100*28*28
#define E0    627200     // N0*8
#define N1    19600      // 100*14*14
#define E1    156800     // N1*8
#define YCOLS 1600       // 25 taps * 64 out channels
#define FCK   3136       // 49*64
#define FCN   512

typedef _Float16 v8h  __attribute__((ext_vector_type(8)));
typedef _Float16 v16h __attribute__((ext_vector_type(16)));
typedef float    v8f  __attribute__((ext_vector_type(8)));

__device__ __forceinline__ float eluf(float v) { return v > 0.f ? v : expm1f(v); }

// ---------------- zero init ----------------
__global__ void k_zero(float* __restrict__ p, int n) {
    int i = blockIdx.x * blockDim.x + threadIdx.x;
    if (i < n) p[i] = 0.f;
}

// ---------------- conv1: edge scatter (Cin=1) ----------------
// thread = (edge, out-channel of 32)
__global__ void k_conv1_scatter(const float* __restrict__ x,
                                const float* __restrict__ pseudo,
                                const int*   __restrict__ eidx,
                                const float* __restrict__ W1,   // [25,32]
                                float* __restrict__ agg,        // [N0,32]
                                float* __restrict__ deg) {      // [N0]
    int gid = blockIdx.x * blockDim.x + threadIdx.x;
    int e = gid >> 5, o = gid & 31;
    if (e >= E0) return;
    float posx = pseudo[2 * e]     * (KTAPS - 1);
    float posy = pseudo[2 * e + 1] * (KTAPS - 1);
    float kfx = floorf(posx), kfy = floorf(posy);
    float fx = posx - kfx, fy = posy - kfy;
    int k0x = min(max((int)kfx, 0), KTAPS - 1);
    int k0y = min(max((int)kfy, 0), KTAPS - 1);
    int k1x = min(k0x + 1, KTAPS - 1);
    int k1y = min(k0y + 1, KTAPS - 1);
    int src = eidx[e], dst = eidx[E0 + e];
    float xs = x[src];
    float w00 = (1.f - fx) * (1.f - fy), w01 = (1.f - fx) * fy;
    float w10 = fx * (1.f - fy),         w11 = fx * fy;
    float v = w00 * W1[(k0x * KTAPS + k0y) * 32 + o]
            + w01 * W1[(k0x * KTAPS + k1y) * 32 + o]
            + w10 * W1[(k1x * KTAPS + k0y) * 32 + o]
            + w11 * W1[(k1x * KTAPS + k1y) * 32 + o];
    atomicAdd(&agg[dst * 32 + o], xs * v);
    if (o == 0) atomicAdd(&deg[dst], 1.f);
}

// ---------------- conv1: finalize + ELU ----------------
__global__ void k_conv1_fin(const float* __restrict__ agg, const float* __restrict__ deg,
                            const float* __restrict__ x,   const float* __restrict__ root1,
                            const float* __restrict__ b1,  float* __restrict__ h1) {
    int gid = blockIdx.x * blockDim.x + threadIdx.x;  // N0*32
    int n = gid >> 5, o = gid & 31;
    if (n >= N0) return;
    float v = agg[gid] / fmaxf(deg[n], 1.f) + x[n] * root1[o] + b1[o];
    h1[gid] = eluf(v);
}

// ---------------- pool1: (100,28,28,32) -> (100,14,14,32), write f32 + f16 ----------------
__global__ void k_pool1(const float* __restrict__ h1, float* __restrict__ p1,
                        _Float16* __restrict__ p1h) {
    int gid = blockIdx.x * blockDim.x + threadIdx.x;  // 100*14*14*32
    if (gid >= N1 * 32) return;
    int c = gid & 31, rest = gid >> 5;
    int j2 = rest % 14; rest /= 14;
    int i2 = rest % 14; int b = rest / 14;
    float m = -3.4e38f;
    for (int di = 0; di < 2; ++di)
        for (int dj = 0; dj < 2; ++dj) {
            int n = b * 784 + (2 * i2 + di) * 28 + (2 * j2 + dj);
            m = fmaxf(m, h1[n * 32 + c]);
        }
    p1[gid]  = m;
    p1h[gid] = (_Float16)m;
}

// ---------------- build W2^T in f16: w2t[col][i], col = k*64+o ----------------
__global__ void k_w2t(const float* __restrict__ W2, _Float16* __restrict__ w2t) {
    int gid = blockIdx.x * blockDim.x + threadIdx.x;  // 1600*32
    if (gid >= YCOLS * 32) return;
    int col = gid >> 5, i = gid & 31;
    int k = col >> 6, o = col & 63;
    w2t[col * 32 + i] = (_Float16)W2[k * 2048 + i * 64 + o];
}

// ---------------- conv2 tap-feature precompute: y2 = p1 @ W2flat (WMMA) ----------------
// One wave per 16x32 output strip: A tile loaded once, reused for 2 N-tiles.
__global__ void k_y2_wmma(const _Float16* __restrict__ A,   // [N1,32]
                          const _Float16* __restrict__ Bt,  // [1600,32] (col-major B)
                          _Float16* __restrict__ Y) {       // [N1,1600] f16
    const int MT = N1 / 16, NT2 = YCOLS / 32;   // 1225 x 50 strips
    int wid = (blockIdx.x * blockDim.x + threadIdx.x) >> 5;
    if (wid >= MT * NT2) return;
    int mt = wid / NT2, ns = wid % NT2;
    int lane = threadIdx.x & 31, half = lane >> 4, l16 = lane & 15;
    // A layout: lane half hw holds K = hw*16+i (contiguous, matching B)
    const _Float16* ap = A + (mt * 16 + l16) * 32;
    v8h a0 = *(const v8h*)(ap + half * 16);
    v8h a1 = *(const v8h*)(ap + half * 16 + 8);
    v16h a;
#pragma unroll
    for (int i = 0; i < 8; ++i) { a[i] = a0[i]; a[i + 8] = a1[i]; }
    int rbase = mt * 16 + half * 8;
#pragma unroll
    for (int s = 0; s < 2; ++s) {
        int nt = ns * 2 + s;
        // B layout: lanes 0-15 hold K=0..15 of col n, lanes 16-31 hold K=16..31 (contiguous in Bt)
        v16h b = *(const v16h*)(Bt + (nt * 16 + l16) * 32 + half * 16);
        v8f c = {};
        c = __builtin_amdgcn_wmma_f32_16x16x32_f16(false, a, false, b, (short)0, c, false, false);
        int col = nt * 16 + l16;
#pragma unroll
        for (int r = 0; r < 8; ++r)
            Y[(rbase + r) * YCOLS + col] = (_Float16)c[r];
    }
}

// ---------------- conv2: edge scatter over y2 ----------------
__global__ void k_conv2_scatter(const _Float16* __restrict__ Y,
                                const float* __restrict__ pseudo,
                                const int*   __restrict__ eidx,
                                float* __restrict__ agg,   // [N1,64]
                                float* __restrict__ deg) { // [N1]
    int gid = blockIdx.x * blockDim.x + threadIdx.x;  // E1*64
    int e = gid >> 6, o = gid & 63;
    if (e >= E1) return;
    float posx = pseudo[2 * e]     * (KTAPS - 1);
    float posy = pseudo[2 * e + 1] * (KTAPS - 1);
    float kfx = floorf(posx), kfy = floorf(posy);
    float fx = posx - kfx, fy = posy - kfy;
    int k0x = min(max((int)kfx, 0), KTAPS - 1);
    int k0y = min(max((int)kfy, 0), KTAPS - 1);
    int k1x = min(k0x + 1, KTAPS - 1);
    int k1y = min(k0y + 1, KTAPS - 1);
    int src = eidx[e], dst = eidx[E1 + e];
    const _Float16* yr = Y + src * YCOLS;
    float w00 = (1.f - fx) * (1.f - fy), w01 = (1.f - fx) * fy;
    float w10 = fx * (1.f - fy),         w11 = fx * fy;
    float v = w00 * (float)yr[(k0x * KTAPS + k0y) * 64 + o]
            + w01 * (float)yr[(k0x * KTAPS + k1y) * 64 + o]
            + w10 * (float)yr[(k1x * KTAPS + k0y) * 64 + o]
            + w11 * (float)yr[(k1x * KTAPS + k1y) * 64 + o];
    atomicAdd(&agg[dst * 64 + o], v);
    if (o == 0) atomicAdd(&deg[dst], 1.f);
}

// ---------------- conv2: finalize (root term) + ELU ----------------
__global__ void k_conv2_fin(const float* __restrict__ agg, const float* __restrict__ deg,
                            const float* __restrict__ p1,  const float* __restrict__ root2,
                            const float* __restrict__ b2,  float* __restrict__ h2) {
    int gid = blockIdx.x * blockDim.x + threadIdx.x;  // N1*64
    int n = gid >> 6, o = gid & 63;
    if (n >= N1) return;
    float acc = agg[gid] / fmaxf(deg[n], 1.f);
    const float* pr = p1 + n * 32;
    float d = 0.f;
#pragma unroll
    for (int i = 0; i < 32; ++i) d += pr[i] * root2[i * 64 + o];
    h2[gid] = eluf(acc + d + b2[o]);
}

// ---------------- pool2: reference reshapes to (25,28,28,64)! -> (25,14,14,64) -> [100,3136] f16 ----------------
__global__ void k_pool2(const float* __restrict__ h2, _Float16* __restrict__ p2h) {
    int gid = blockIdx.x * blockDim.x + threadIdx.x;  // 25*14*14*64 == 100*3136
    if (gid >= NB * FCK) return;
    int c = gid & 63, rest = gid >> 6;
    int j2 = rest % 14; rest /= 14;
    int i2 = rest % 14; int b2 = rest / 14;  // b2 in [0,25)
    float m = -3.4e38f;
    for (int di = 0; di < 2; ++di)
        for (int dj = 0; dj < 2; ++dj) {
            int n = b2 * 784 + (2 * i2 + di) * 28 + (2 * j2 + dj);
            m = fmaxf(m, h2[n * 64 + c]);
        }
    p2h[gid] = (_Float16)m;  // gid == row-major flatten of (b2,i2,j2,c) == [100,3136]
}

// ---------------- build fc1_w^T in f16: [512][3136] ----------------
__global__ void k_fc1wt(const float* __restrict__ W, _Float16* __restrict__ Wt) {
    int gid = blockIdx.x * blockDim.x + threadIdx.x;  // 512*3136
    if (gid >= FCN * FCK) return;
    int n = gid / FCK, k = gid % FCK;
    Wt[n * FCK + k] = (_Float16)W[k * FCN + n];
}

// ---------------- fc1: [112,3136]x[3136,512] WMMA, fused bias+ELU ----------------
// A is pre-padded to 112 rows (rows 100..111 zeroed once) -> branch-free K-loop.
__global__ void k_fc1_wmma(const _Float16* __restrict__ A,   // [112,3136]
                           const _Float16* __restrict__ Bt,  // [512,3136]
                           const float* __restrict__ bias,
                           float* __restrict__ O) {          // [100,512]
    int wid = (blockIdx.x * blockDim.x + threadIdx.x) >> 5;  // 224 waves exactly
    int nt = wid & 31, mt = wid >> 5;                        // 32 N-tiles, 7 M-tiles
    int lane = threadIdx.x & 31, half = lane >> 4, l16 = lane & 15;
    const _Float16* ap = A + (mt * 16 + l16) * FCK + half * 16;
    const _Float16* bp = Bt + (nt * 16 + l16) * FCK + half * 16;
    v8f c = {};
#pragma unroll 2
    for (int kk = 0; kk < FCK / 32; ++kk) {
        v8h a0 = *(const v8h*)(ap + kk * 32);
        v8h a1 = *(const v8h*)(ap + kk * 32 + 8);
        v16h a;
#pragma unroll
        for (int i = 0; i < 8; ++i) { a[i] = a0[i]; a[i + 8] = a1[i]; }
        v16h b = *(const v16h*)(bp + kk * 32);
        c = __builtin_amdgcn_wmma_f32_16x16x32_f16(false, a, false, b, (short)0, c, false, false);
    }
    int col = nt * 16 + l16;
    float bb = bias[col];
    int rbase = mt * 16 + half * 8;
#pragma unroll
    for (int r = 0; r < 8; ++r) {
        int row = rbase + r;
        if (row < NB) O[row * FCN + col] = eluf(c[r] + bb);
    }
}

// ---------------- fc2 + ELU + log_softmax ----------------
__global__ void k_fc2_lsm(const float* __restrict__ H, const float* __restrict__ W,
                          const float* __restrict__ b, float* __restrict__ out) {
    int row = blockIdx.x;
    int t = threadIdx.x;
    __shared__ float l[10];
    __shared__ float lse;
    if (t < 10) {
        float acc = b[t];
        const float* h = H + row * FCN;
        for (int k = 0; k < FCN; ++k) acc += h[k] * W[k * 10 + t];
        l[t] = eluf(acc);
    }
    __syncthreads();
    if (t == 0) {
        float m = l[0];
        for (int i = 1; i < 10; ++i) m = fmaxf(m, l[i]);
        float s = 0.f;
        for (int i = 0; i < 10; ++i) s += expf(l[i] - m);
        lse = m + logf(s);
    }
    __syncthreads();
    if (t < 10) out[row * 10 + t] = l[t] - lse;
}

// ---------------- launcher ----------------
extern "C" void kernel_launch(void* const* d_in, const int* in_sizes, int n_in,
                              void* d_out, int out_size, void* d_ws, size_t ws_size,
                              hipStream_t stream) {
    const float* x      = (const float*)d_in[0];
    const float* ps0    = (const float*)d_in[1];
    const float* ps1    = (const float*)d_in[2];
    const float* W1     = (const float*)d_in[3];
    const float* root1  = (const float*)d_in[4];
    const float* b1     = (const float*)d_in[5];
    const float* W2     = (const float*)d_in[6];
    const float* root2  = (const float*)d_in[7];
    const float* b2     = (const float*)d_in[8];
    const float* fc1_w  = (const float*)d_in[9];
    const float* fc1_b  = (const float*)d_in[10];
    const float* fc2_w  = (const float*)d_in[11];
    const float* fc2_b  = (const float*)d_in[12];
    const int*   e0     = (const int*)d_in[13];
    const int*   e1     = (const int*)d_in[14];
    float* out = (float*)d_out;

    // workspace carve (256B aligned regions); accumulators first so one zero pass covers them
    size_t off = 0;
    char* base = (char*)d_ws;
    auto carve = [&](size_t bytes) -> void* {
        void* p = base + off;
        off += (bytes + 255) & ~(size_t)255;
        return p;
    };
    float*    agg0  = (float*)carve((size_t)N0 * 32 * 4);
    float*    deg0  = (float*)carve((size_t)N0 * 4);
    float*    agg1  = (float*)carve((size_t)N1 * 64 * 4);
    float*    deg1  = (float*)carve((size_t)N1 * 4);
    size_t zero_bytes = off;               // contiguous accumulator span
    float*    h1    = (float*)carve((size_t)N0 * 32 * 4);
    float*    p1    = (float*)carve((size_t)N1 * 32 * 4);
    _Float16* p1h   = (_Float16*)carve((size_t)N1 * 32 * 2);
    _Float16* w2t   = (_Float16*)carve((size_t)YCOLS * 32 * 2);
    _Float16* y2    = (_Float16*)carve((size_t)N1 * YCOLS * 2);   // 62.7 MB, lives in L2
    float*    h2    = (float*)carve((size_t)N1 * 64 * 4);
    _Float16* p2h   = (_Float16*)carve((size_t)NBPAD * FCK * 2);  // padded to 112 rows
    _Float16* fc1wt = (_Float16*)carve((size_t)FCK * FCN * 2);
    float*    fc1o  = (float*)carve((size_t)NB * FCN * 4);

    const int T = 256;
    int zn = (int)(zero_bytes / 4);
    k_zero<<<(zn + T - 1) / T, T, 0, stream>>>(agg0, zn);
    // zero the 12 pad rows of the fc1 activation matrix (f16, 2 halves per float slot)
    int zp = (NBPAD - NB) * FCK / 2;       // 18816 floats
    k_zero<<<(zp + T - 1) / T, T, 0, stream>>>((float*)(p2h + (size_t)NB * FCK), zp);

    k_conv1_scatter<<<(E0 * 32) / T, T, 0, stream>>>(x, ps0, e0, W1, agg0, deg0);
    k_conv1_fin<<<(N0 * 32) / T, T, 0, stream>>>(agg0, deg0, x, root1, b1, h1);
    k_pool1<<<(N1 * 32) / T, T, 0, stream>>>(h1, p1, p1h);

    k_w2t<<<(YCOLS * 32) / T, T, 0, stream>>>(W2, w2t);
    {
        int waves = (N1 / 16) * (YCOLS / 32);   // 61250 strips (2 WMMA tiles each)
        int blocks = (waves * 32 + T - 1) / T;
        k_y2_wmma<<<blocks, T, 0, stream>>>(p1h, w2t, y2);
    }

    k_conv2_scatter<<<(E1 * 64) / T, T, 0, stream>>>(y2, ps1, e1, agg1, deg1);
    k_conv2_fin<<<(N1 * 64) / T, T, 0, stream>>>(agg1, deg1, p1, root2, b2, h2);
    k_pool2<<<(NB * FCK) / T, T, 0, stream>>>(h2, p2h);

    k_fc1wt<<<(FCN * FCK) / T, T, 0, stream>>>(fc1_w, fc1wt);
    k_fc1_wmma<<<28, T, 0, stream>>>(p2h, fc1wt, fc1_b, fc1o);   // 7*32 = 224 waves exactly

    k_fc2_lsm<<<NB, 32, 0, stream>>>(fc1o, fc2_w, fc2_b, out);
}
